// InvariantPointAttention_481036337750
// MI455X (gfx1250) — matrix operand order repaired and checked by reading, not compiled, not run on hardware
//
#include <hip/hip_runtime.h>
#include <hip/hip_bf16.h>

// ---------------------------------------------------------------------------
// Types for CDNA5 WMMA (wave32): 16x16x32 bf16 -> f32
// ---------------------------------------------------------------------------
typedef __attribute__((ext_vector_type(16))) __bf16        v16bf;
typedef __attribute__((ext_vector_type(8)))  float         v8f;
typedef __attribute__((ext_vector_type(8)))  unsigned int  v8u;

// Problem constants
#define LSEQ 512
#define DSNG 384
#define DPAIR 128
#define NH 12
#define DH 32
#define PQ 4
#define PV 8
#define OUTD 2304   // H*DH + H*PV*4 + H*DPAIR = 384+384+1536

__device__ __forceinline__ unsigned short f2bf(float f) {
    unsigned int u = __builtin_bit_cast(unsigned int, f);
    unsigned int r = u + 0x7fffu + ((u >> 16) & 1u);   // round-to-nearest-even
    return (unsigned short)(r >> 16);
}
__device__ __forceinline__ unsigned int pack2(unsigned short lo, unsigned short hi) {
    return (unsigned int)lo | ((unsigned int)hi << 16);
}

__device__ __forceinline__ v8f wmma_bf(v16bf a, v16bf b, v8f c) {
    // v_wmma_f32_16x16x32_bf16  (neg_a, A, neg_b, B, c_mod, C, reuse_a, reuse_b)
    return __builtin_amdgcn_wmma_f32_16x16x32_bf16(false, a, false, b, (short)0, c, false, false);
}

// A-operand layout (16-bit A 16x32): lane m=lane&15, half=lane>>4.
// slot j : K = half*16 + 2j (contiguous, matches B)
__device__ __forceinline__ v16bf load_a_bf16(const unsigned short* p, int half, bool ok) {
    v8u u = {0u,0u,0u,0u,0u,0u,0u,0u};
    if (ok) {
        int kb = half * 16;
#pragma unroll
        for (int j = 0; j < 4; ++j) u[j]     = *(const unsigned int*)(p + kb + 2*j);
#pragma unroll
        for (int j = 0; j < 4; ++j) u[4 + j] = *(const unsigned int*)(p + 8 + kb + 2*j);
    }
    return __builtin_bit_cast(v16bf, u);
}
__device__ __forceinline__ v16bf load_a_bf16_k(const unsigned short* p, int half, int kmax) {
    v8u u = {0u,0u,0u,0u,0u,0u,0u,0u};
    int kb = half * 16;
#pragma unroll
    for (int j = 0; j < 4; ++j) { int k = kb + 2*j;      if (k + 1 < kmax) u[j]     = *(const unsigned int*)(p + k); }
#pragma unroll
    for (int j = 0; j < 4; ++j) { int k = 8 + kb + 2*j; if (k + 1 < kmax) u[4 + j] = *(const unsigned int*)(p + k); }
    return __builtin_bit_cast(v16bf, u);
}
__device__ __forceinline__ v16bf load_a_f32(const float* p, int half, bool ok) {
    v8u u = {0u,0u,0u,0u,0u,0u,0u,0u};
    if (ok) {
        int kb = half * 16;
#pragma unroll
        for (int j = 0; j < 4; ++j) { int k = kb + 2*j;      u[j]     = pack2(f2bf(p[k]), f2bf(p[k+1])); }
#pragma unroll
        for (int j = 0; j < 4; ++j) { int k = 8 + kb + 2*j; u[4 + j] = pack2(f2bf(p[k]), f2bf(p[k+1])); }
    }
    return __builtin_bit_cast(v16bf, u);
}
// B-operand layout (16-bit B 32x16): lane n=lane&15, half=lane>>4, slot j: K = half*16 + 2j.
// Bt is B transposed row-major [N x K] so K pairs are contiguous.
__device__ __forceinline__ v16bf load_bt_bf16(const unsigned short* p, int half, bool ok) {
    v8u u = {0u,0u,0u,0u,0u,0u,0u,0u};
    if (ok) {
        int kb = half * 16;
#pragma unroll
        for (int j = 0; j < 8; ++j) u[j] = *(const unsigned int*)(p + kb + 2*j);
    }
    return __builtin_bit_cast(v16bf, u);
}
__device__ __forceinline__ v16bf load_bt_bf16_k(const unsigned short* p, int half, int kmax) {
    v8u u = {0u,0u,0u,0u,0u,0u,0u,0u};
    int kb = half * 16;
#pragma unroll
    for (int j = 0; j < 8; ++j) { int k = kb + 2*j; if (k + 1 < kmax) u[j] = *(const unsigned int*)(p + k); }
    return __builtin_bit_cast(v16bf, u);
}
// B normal-major f32 [K x N]: element (k, n) at p0[k*ldb + n]; pass p = &B[kb*ldb + n]
__device__ __forceinline__ v16bf load_bn_f32(const float* p, long ldb, int half, bool ok) {
    v8u u = {0u,0u,0u,0u,0u,0u,0u,0u};
    if (ok) {
        int kb = half * 16;
#pragma unroll
        for (int j = 0; j < 8; ++j) {
            long k = kb + 2*j;
            u[j] = pack2(f2bf(p[k * ldb]), f2bf(p[(k + 1) * ldb]));
        }
    }
    return __builtin_bit_cast(v16bf, u);
}

// ---------------------------------------------------------------------------
// Generic tiled WMMA GEMM: C[M,N] = A[M,K] * B[K,N] (+bias[n]).
// One wave computes one 16x16 tile. K must be a multiple of 32; M,N tails guarded.
// AF32:  A is f32 (converted on the fly), else bf16 (row-major, lda elems).
// BNF32: B is f32 normal-major [KxN] (ldb), else bf16 transposed [NxK] (ldb = ldbt).
// OUTBF: store bf16 into Cb, else f32 into Cf.
// ---------------------------------------------------------------------------
template<bool AF32, bool BNF32, bool OUTBF>
__global__ void gemm_wmma_kernel(const void* __restrict__ Ap, long lda,
                                 const void* __restrict__ Bp, long ldb,
                                 float* __restrict__ Cf, unsigned short* __restrict__ Cb,
                                 long ldc, const float* __restrict__ bias,
                                 int M, int N, int K, int tilesN, long tiles)
{
    int wave = threadIdx.x >> 5;
    int lane = threadIdx.x & 31;
    long gw = (long)blockIdx.x * (blockDim.x >> 5) + wave;
    if (gw >= tiles) return;                 // wave-uniform: EXEC stays all-ones
    int tm = (int)(gw / tilesN);
    int tn = (int)(gw % tilesN);
    int half = lane >> 4, l16 = lane & 15;
    long arow = (long)tm * 16 + l16;
    long bcol = (long)tn * 16 + l16;
    bool aok = arow < M;
    bool bok = bcol < N;
    const float*          Af = (const float*)Ap;
    const unsigned short* Ab = (const unsigned short*)Ap;
    const float*          Bf = (const float*)Bp;
    const unsigned short* Bb = (const unsigned short*)Bp;

    v8f acc = {0.f,0.f,0.f,0.f,0.f,0.f,0.f,0.f};
    for (int kb = 0; kb < K; kb += 32) {
        v16bf a, b;
        if (AF32) {
            if (aok && kb + 32 < K) __builtin_prefetch(Af + arow * lda + kb + 32, 0, 1);
            a = load_a_f32(Af + arow * lda + kb, half, aok);
        } else {
            a = load_a_bf16(Ab + arow * lda + kb, half, aok);
        }
        if (BNF32) {
            if (bok && kb + 32 < K) __builtin_prefetch(Bf + (long)(kb + 32) * ldb + bcol, 0, 1);
            b = load_bn_f32(Bf + (long)kb * ldb + bcol, ldb, half, bok);
        } else {
            b = load_bt_bf16(Bb + bcol * ldb + kb, half, bok);
        }
        acc = wmma_bf(a, b, acc);
    }
    int n = tn * 16 + l16;
    if (n < N) {
        float bv = bias ? bias[n] : 0.0f;
#pragma unroll
        for (int r = 0; r < 8; ++r) {
            long m = (long)tm * 16 + half * 8 + r;
            if (m < M) {
                float val = acc[r] + bv;
                if (OUTBF) Cb[m * ldc + n] = f2bf(val);
                else       Cf[m * ldc + n] = val;
            }
        }
    }
}

// ---------------------------------------------------------------------------
// LayerNorm: one block (128 thr) per row of single [512 x 384] -> bf16
// ---------------------------------------------------------------------------
__global__ void ln_kernel(const float* __restrict__ x, const float* __restrict__ w,
                          const float* __restrict__ b, unsigned short* __restrict__ out)
{
    int l = blockIdx.x;
    const float* row = x + (long)l * DSNG;
    __shared__ float red[4];
    __shared__ float mu_s, rs_s;
    int t = threadIdx.x, wid = t >> 5, ln = t & 31;

    float s = 0.f;
    for (int d = t; d < DSNG; d += blockDim.x) s += row[d];
    for (int o = 16; o; o >>= 1) s += __shfl_xor(s, o, 32);
    if (ln == 0) red[wid] = s;
    __syncthreads();
    if (t == 0) mu_s = (red[0] + red[1] + red[2] + red[3]) * (1.0f / DSNG);
    __syncthreads();
    float mu = mu_s;

    float vs = 0.f;
    for (int d = t; d < DSNG; d += blockDim.x) { float dv = row[d] - mu; vs += dv * dv; }
    for (int o = 16; o; o >>= 1) vs += __shfl_xor(vs, o, 32);
    __syncthreads();
    if (ln == 0) red[wid] = vs;
    __syncthreads();
    if (t == 0) rs_s = rsqrtf((red[0] + red[1] + red[2] + red[3]) * (1.0f / DSNG) + 1e-5f);
    __syncthreads();
    float rs = rs_s;
    for (int d = t; d < DSNG; d += blockDim.x)
        out[(long)l * DSNG + d] = f2bf((row[d] - mu) * rs * w[d] + b[d]);
}

// Weight transpose + f32->bf16: WT[n*K + k] = W[k*N + n]
__global__ void wconv_t_kernel(const float* __restrict__ W, unsigned short* __restrict__ WT,
                               int K, int N)
{
    long gid = (long)blockIdx.x * blockDim.x + threadIdx.x;
    if (gid >= (long)K * N) return;
    long k = gid / N, n = gid % N;
    WT[n * (long)K + k] = f2bf(W[k * (long)N + n]);
}

// bf16 transpose for V: vT[d*512 + j] = v[j*384 + d]
__global__ void vtrans_kernel(const unsigned short* __restrict__ v, unsigned short* __restrict__ vT)
{
    long gid = (long)blockIdx.x * blockDim.x + threadIdx.x;
    if (gid >= (long)DSNG * LSEQ) return;
    long d = gid / LSEQ, j = gid % LSEQ;
    vT[d * LSEQ + j] = v[j * DSNG + d];
}

// ---------------------------------------------------------------------------
// Rigid transforms: qg = R*qp + t, kg = R*kp + t, vg = R*vp + t.
// Writes qg/kg bf16 [512 x 144] (12 dims per head), vg^T bf16 [288 x 512],
// and row sums Qsq/Ksq [512 x 12] for the distance identity.
// ---------------------------------------------------------------------------
__global__ void rigid_kernel(const float* __restrict__ qp, const float* __restrict__ kp,
                             const float* __restrict__ vp, const float* __restrict__ rot,
                             const float* __restrict__ trans,
                             unsigned short* __restrict__ qg, unsigned short* __restrict__ kg,
                             unsigned short* __restrict__ vgT,
                             float* __restrict__ Qsq, float* __restrict__ Ksq)
{
    int gid = blockIdx.x * blockDim.x + threadIdx.x;
    if (gid >= LSEQ * NH) return;
    int l = gid / NH, h = gid % NH;
    float R[9];
#pragma unroll
    for (int i = 0; i < 9; ++i) R[i] = rot[(long)l * 9 + i];
    float t0 = trans[l * 3 + 0], t1 = trans[l * 3 + 1], t2 = trans[l * 3 + 2];
    float qs = 0.f, ks = 0.f;
#pragma unroll
    for (int p = 0; p < PQ; ++p) {
        const float* q = qp + (long)l * (NH * PQ * 3) + h * (PQ * 3) + p * 3;
        float g0 = R[0]*q[0] + R[1]*q[1] + R[2]*q[2] + t0;
        float g1 = R[3]*q[0] + R[4]*q[1] + R[5]*q[2] + t1;
        float g2 = R[6]*q[0] + R[7]*q[1] + R[8]*q[2] + t2;
        qs += g0*g0 + g1*g1 + g2*g2;
        unsigned short* o = qg + (long)l * (NH * 12) + h * 12 + p * 3;
        o[0] = f2bf(g0); o[1] = f2bf(g1); o[2] = f2bf(g2);

        const float* kq = kp + (long)l * (NH * PQ * 3) + h * (PQ * 3) + p * 3;
        float h0 = R[0]*kq[0] + R[1]*kq[1] + R[2]*kq[2] + t0;
        float h1 = R[3]*kq[0] + R[4]*kq[1] + R[5]*kq[2] + t1;
        float h2 = R[6]*kq[0] + R[7]*kq[1] + R[8]*kq[2] + t2;
        ks += h0*h0 + h1*h1 + h2*h2;
        unsigned short* ok = kg + (long)l * (NH * 12) + h * 12 + p * 3;
        ok[0] = f2bf(h0); ok[1] = f2bf(h1); ok[2] = f2bf(h2);
    }
#pragma unroll
    for (int p = 0; p < PV; ++p) {
        const float* v = vp + (long)l * (NH * PV * 3) + h * (PV * 3) + p * 3;
        float g0 = R[0]*v[0] + R[1]*v[1] + R[2]*v[2] + t0;
        float g1 = R[3]*v[0] + R[4]*v[1] + R[5]*v[2] + t1;
        float g2 = R[6]*v[0] + R[7]*v[1] + R[8]*v[2] + t2;
        long base = (long)(h * (PV * 3) + p * 3) * LSEQ + l;
        vgT[base + 0 * LSEQ] = f2bf(g0);
        vgT[base + 1 * LSEQ] = f2bf(g1);
        vgT[base + 2 * LSEQ] = f2bf(g2);
    }
    Qsq[l * NH + h] = qs;
    Ksq[l * NH + h] = ks;
}

// ---------------------------------------------------------------------------
// Fused logits: seq (QK^T wmma, K=32) + geometry (wmma over 12 dims padded
// to 32, using |q|^2+|k|^2-2 q.k) + pair bias + mask.
// grid (8, 32, 12), block 128 (4 waves, one 16x16 j-tile each).
// ---------------------------------------------------------------------------
__global__ void logits_kernel(const unsigned short* __restrict__ qb,
                              const unsigned short* __restrict__ kb16,
                              const unsigned short* __restrict__ qgb,
                              const unsigned short* __restrict__ kgb,
                              const float* __restrict__ Qsq, const float* __restrict__ Ksq,
                              const float* __restrict__ pb, const float* __restrict__ mask,
                              const float* __restrict__ wc_p, const float* __restrict__ wl_p,
                              const float* __restrict__ gamma, float* __restrict__ logits)
{
    int h = blockIdx.z, tm = blockIdx.y;
    int wave = threadIdx.x >> 5, lane = threadIdx.x & 31;
    int tn = blockIdx.x * 4 + wave;
    int half = lane >> 4, l16 = lane & 15;
    int irow = tm * 16 + l16;   // A-operand row (query index)
    int jrow = tn * 16 + l16;   // B-operand column (key index)

    v8f aqk = {0.f,0.f,0.f,0.f,0.f,0.f,0.f,0.f};
    v8f ag  = {0.f,0.f,0.f,0.f,0.f,0.f,0.f,0.f};

    v16bf a  = load_a_bf16 (qb  + (long)irow * DSNG + h * DH, half, true);
    v16bf b  = load_bt_bf16(kb16 + (long)jrow * DSNG + h * DH, half, true);
    aqk = wmma_bf(a, b, aqk);

    v16bf a2 = load_a_bf16_k (qgb + (long)irow * (NH * 12) + h * 12, half, 12);
    v16bf b2 = load_bt_bf16_k(kgb + (long)jrow * (NH * 12) + h * 12, half, 12);
    ag = wmma_bf(a2, b2, ag);

    float wc = wc_p[0], wl = wl_p[0], g = gamma[h];
    float sscale = wc * 0.17677669529663687f;   // 1/sqrt(DH)
    float gscale = -0.5f * g * wl;
    int j = tn * 16 + l16;
    float mj = mask[j];
#pragma unroll
    for (int r = 0; r < 8; ++r) {
        int i = tm * 16 + half * 8 + r;
        float val = aqk[r] * sscale
                  + gscale * (Qsq[i * NH + h] + Ksq[j * NH + h] - 2.0f * ag[r])
                  + pb[((long)i * LSEQ + j) * NH + h];
        if (!(mask[i] * mj > 0.f)) val = -1.0e9f;
        logits[((long)h * LSEQ + i) * LSEQ + j] = val;
    }
}

// Softmax over j (512): one block (256 thr) per (h,i) row -> attn bf16
__global__ void softmax_kernel(const float* __restrict__ logits, unsigned short* __restrict__ attn)
{
    long row = blockIdx.x;
    const float* src = logits + row * LSEQ;
    __shared__ float red_max[8], red_sum[8];
    int t = threadIdx.x, wid = t >> 5, ln = t & 31;

    float v0 = src[t], v1 = src[t + 256];
    float m = fmaxf(v0, v1);
    for (int o = 16; o; o >>= 1) m = fmaxf(m, __shfl_xor(m, o, 32));
    if (ln == 0) red_max[wid] = m;
    __syncthreads();
    float M = red_max[0];
#pragma unroll
    for (int q = 1; q < 8; ++q) M = fmaxf(M, red_max[q]);

    float e0 = __expf(v0 - M), e1 = __expf(v1 - M);
    float s = e0 + e1;
    for (int o = 16; o; o >>= 1) s += __shfl_xor(s, o, 32);
    if (ln == 0) red_sum[wid] = s;
    __syncthreads();
    float S = red_sum[0];
#pragma unroll
    for (int q = 1; q < 8; ++q) S += red_sum[q];
    float inv = 1.0f / S;
    attn[row * LSEQ + t]       = f2bf(e0 * inv);
    attn[row * LSEQ + t + 256] = f2bf(e1 * inv);
}

// ---------------------------------------------------------------------------
// pair_out[i,h,d] = sum_j attn[h,i,j] * pair[i,j,d]  — 512 independent
// (12x16-padded) x 128 x 512 GEMMs. One block per i, 8 waves = 8 N-tiles.
// pair is read once, f32, converted on the fly.
// ---------------------------------------------------------------------------
__global__ void pair_out_kernel(const unsigned short* __restrict__ attn,
                                const float* __restrict__ pair,
                                unsigned short* __restrict__ combined)
{
    int i = blockIdx.x;
    int wave = threadIdx.x >> 5, lane = threadIdx.x & 31;
    int half = lane >> 4, l16 = lane & 15;
    int m = l16;                   // head index (rows 12..15 are padding)
    bool mok = m < NH;
    int n = wave * 16 + l16;       // pair channel 0..127

    const unsigned short* Abase = attn + (long)m * (LSEQ * LSEQ) + (long)i * LSEQ;
    const float*          Bbase = pair + (long)i * LSEQ * DPAIR + n;

    v8f acc = {0.f,0.f,0.f,0.f,0.f,0.f,0.f,0.f};
    for (int kb = 0; kb < LSEQ; kb += 32) {
        if (kb + 32 < LSEQ) __builtin_prefetch(Bbase + (long)(kb + 32) * DPAIR, 0, 1);
        v16bf a = load_a_bf16(Abase + kb, half, mok);
        v16bf b = load_bn_f32(Bbase + (long)kb * DPAIR, DPAIR, half, true);
        acc = wmma_bf(a, b, acc);
    }
    unsigned short* C = combined + (long)i * OUTD + 768;  // pair_out section
    int cn = wave * 16 + l16;
#pragma unroll
    for (int r = 0; r < 8; ++r) {
        int mm = half * 8 + r;
        if (mm < NH) C[(long)mm * DPAIR + cn] = f2bf(acc[r]);
    }
}

// Point post-process: local = R^T (pts - t), feat = [norm, local] -> combined bf16
__global__ void pts_feat_kernel(const float* __restrict__ pts, const float* __restrict__ rot,
                                const float* __restrict__ trans, unsigned short* __restrict__ combined)
{
    int gid = blockIdx.x * blockDim.x + threadIdx.x;
    if (gid >= LSEQ * NH * PV) return;
    int i = gid / (NH * PV), hp = gid % (NH * PV);
    const float* o = pts + (long)i * (NH * PV * 3) + hp * 3;
    float x0 = o[0] - trans[i * 3 + 0];
    float x1 = o[1] - trans[i * 3 + 1];
    float x2 = o[2] - trans[i * 3 + 2];
    const float* R = rot + (long)i * 9;
    float l0 = R[0]*x0 + R[3]*x1 + R[6]*x2;
    float l1 = R[1]*x0 + R[4]*x1 + R[7]*x2;
    float l2 = R[2]*x0 + R[5]*x1 + R[8]*x2;
    float nrm = sqrtf(l0*l0 + l1*l1 + l2*l2);
    unsigned short* c = combined + (long)i * OUTD + 384 + hp * 4;
    c[0] = f2bf(nrm); c[1] = f2bf(l0); c[2] = f2bf(l1); c[3] = f2bf(l2);
}

// ---------------------------------------------------------------------------
// Host-side launch helpers
// ---------------------------------------------------------------------------
static void gemm_AbfBt(hipStream_t st, const unsigned short* A, long lda,
                       const unsigned short* Bt, long ldbt,
                       float* Cf, unsigned short* Cb, long ldc, const float* bias,
                       int M, int N, int K)
{
    int tilesM = (M + 15) / 16, tilesN = (N + 15) / 16;
    long tiles = (long)tilesM * tilesN;
    int blocks = (int)((tiles + 7) / 8);
    if (Cb)
        gemm_wmma_kernel<false,false,true><<<blocks, 256, 0, st>>>(
            (const void*)A, lda, (const void*)Bt, ldbt, nullptr, Cb, ldc, bias, M, N, K, tilesN, tiles);
    else
        gemm_wmma_kernel<false,false,false><<<blocks, 256, 0, st>>>(
            (const void*)A, lda, (const void*)Bt, ldbt, Cf, nullptr, ldc, bias, M, N, K, tilesN, tiles);
}
static void gemm_Af32Bt(hipStream_t st, const float* A, long lda,
                        const unsigned short* Bt, long ldbt,
                        float* Cf, long ldc, const float* bias, int M, int N, int K)
{
    int tilesM = (M + 15) / 16, tilesN = (N + 15) / 16;
    long tiles = (long)tilesM * tilesN;
    int blocks = (int)((tiles + 7) / 8);
    gemm_wmma_kernel<true,false,false><<<blocks, 256, 0, st>>>(
        (const void*)A, lda, (const void*)Bt, ldbt, Cf, nullptr, ldc, bias, M, N, K, tilesN, tiles);
}

extern "C" void kernel_launch(void* const* d_in, const int* in_sizes, int n_in,
                              void* d_out, int out_size, void* d_ws, size_t ws_size,
                              hipStream_t stream)
{
    (void)in_sizes; (void)n_in; (void)out_size; (void)ws_size;
    const float* single  = (const float*)d_in[0];
    const float* pair    = (const float*)d_in[1];
    const float* rot     = (const float*)d_in[2];
    const float* trans   = (const float*)d_in[3];
    const float* mask    = (const float*)d_in[4];
    const float* ln_w    = (const float*)d_in[5];
    const float* ln_b    = (const float*)d_in[6];
    const float* Wq      = (const float*)d_in[7];
    const float* bq      = (const float*)d_in[8];
    const float* Wk      = (const float*)d_in[9];
    const float* bk      = (const float*)d_in[10];
    const float* Wv      = (const float*)d_in[11];
    const float* bv      = (const float*)d_in[12];
    const float* Wqp     = (const float*)d_in[13];
    const float* bqp     = (const float*)d_in[14];
    const float* Wkp     = (const float*)d_in[15];
    const float* bkp     = (const float*)d_in[16];
    const float* Wvp     = (const float*)d_in[17];
    const float* bvp     = (const float*)d_in[18];
    const float* Wpb     = (const float*)d_in[19];
    const float* bpb     = (const float*)d_in[20];
    const float* Wo      = (const float*)d_in[21];
    const float* bo      = (const float*)d_in[22];
    const float* w_c     = (const float*)d_in[23];
    const float* w_l     = (const float*)d_in[24];
    const float* gamma_h = (const float*)d_in[25];
    float* out = (float*)d_out;

    // -------- workspace layout --------
    size_t cur = 0;
    auto alloc = [&](size_t bytes) -> void* {
        void* p = (char*)d_ws + cur;
        cur += (bytes + 255) & ~(size_t)255;
        return p;
    };
    unsigned short* WqT  = (unsigned short*)alloc((size_t)DSNG * 384 * 2);
    unsigned short* WkT  = (unsigned short*)alloc((size_t)DSNG * 384 * 2);
    unsigned short* WvT  = (unsigned short*)alloc((size_t)DSNG * 384 * 2);
    unsigned short* WqpT = (unsigned short*)alloc((size_t)DSNG * 144 * 2);
    unsigned short* WkpT = (unsigned short*)alloc((size_t)DSNG * 144 * 2);
    unsigned short* WvpT = (unsigned short*)alloc((size_t)DSNG * 288 * 2);
    unsigned short* WpbT = (unsigned short*)alloc((size_t)DPAIR * NH * 2);
    unsigned short* WoT  = (unsigned short*)alloc((size_t)OUTD * DSNG * 2);
    unsigned short* sn_bf = (unsigned short*)alloc((size_t)LSEQ * DSNG * 2);
    unsigned short* q_bf  = (unsigned short*)alloc((size_t)LSEQ * DSNG * 2);
    unsigned short* k_bf  = (unsigned short*)alloc((size_t)LSEQ * DSNG * 2);
    unsigned short* v_bf  = (unsigned short*)alloc((size_t)LSEQ * DSNG * 2);
    unsigned short* vT_bf = (unsigned short*)alloc((size_t)DSNG * LSEQ * 2);
    float* qp_f = (float*)alloc((size_t)LSEQ * 144 * 4);
    float* kp_f = (float*)alloc((size_t)LSEQ * 144 * 4);
    float* vp_f = (float*)alloc((size_t)LSEQ * 288 * 4);
    unsigned short* qg_bf  = (unsigned short*)alloc((size_t)LSEQ * NH * 12 * 2);
    unsigned short* kg_bf  = (unsigned short*)alloc((size_t)LSEQ * NH * 12 * 2);
    unsigned short* vgT_bf = (unsigned short*)alloc((size_t)NH * 24 * LSEQ * 2);
    float* Qsq = (float*)alloc((size_t)LSEQ * NH * 4);
    float* Ksq = (float*)alloc((size_t)LSEQ * NH * 4);
    float* pb  = (float*)alloc((size_t)LSEQ * LSEQ * NH * 4);
    float* logits = (float*)alloc((size_t)NH * LSEQ * LSEQ * 4);
    unsigned short* attn_bf = (unsigned short*)alloc((size_t)NH * LSEQ * LSEQ * 2);
    float* pts_f = (float*)alloc((size_t)LSEQ * NH * PV * 3 * 4);
    unsigned short* comb_bf = (unsigned short*)alloc((size_t)LSEQ * OUTD * 2);

    // -------- 1) weight transposes (f32 -> bf16^T) --------
    auto wconv = [&](const float* W, unsigned short* WT, int K, int N) {
        long n = (long)K * N;
        wconv_t_kernel<<<(unsigned)((n + 255) / 256), 256, 0, stream>>>(W, WT, K, N);
    };
    wconv(Wq,  WqT,  DSNG, 384);
    wconv(Wk,  WkT,  DSNG, 384);
    wconv(Wv,  WvT,  DSNG, 384);
    wconv(Wqp, WqpT, DSNG, 144);
    wconv(Wkp, WkpT, DSNG, 144);
    wconv(Wvp, WvpT, DSNG, 288);
    wconv(Wpb, WpbT, DPAIR, NH);
    wconv(Wo,  WoT,  OUTD, DSNG);

    // -------- 2) LayerNorm --------
    ln_kernel<<<LSEQ, 128, 0, stream>>>(single, ln_w, ln_b, sn_bf);

    // -------- 3) projections --------
    gemm_AbfBt(stream, sn_bf, DSNG, WqT,  DSNG, nullptr, q_bf, DSNG, bq,  LSEQ, 384, DSNG);
    gemm_AbfBt(stream, sn_bf, DSNG, WkT,  DSNG, nullptr, k_bf, DSNG, bk,  LSEQ, 384, DSNG);
    gemm_AbfBt(stream, sn_bf, DSNG, WvT,  DSNG, nullptr, v_bf, DSNG, bv,  LSEQ, 384, DSNG);
    gemm_AbfBt(stream, sn_bf, DSNG, WqpT, DSNG, qp_f, nullptr, 144, bqp, LSEQ, 144, DSNG);
    gemm_AbfBt(stream, sn_bf, DSNG, WkpT, DSNG, kp_f, nullptr, 144, bkp, LSEQ, 144, DSNG);
    gemm_AbfBt(stream, sn_bf, DSNG, WvpT, DSNG, vp_f, nullptr, 288, bvp, LSEQ, 288, DSNG);
    vtrans_kernel<<<(unsigned)(((long)DSNG * LSEQ + 255) / 256), 256, 0, stream>>>(v_bf, vT_bf);

    // -------- 4) rigid transforms --------
    rigid_kernel<<<(LSEQ * NH + 255) / 256, 256, 0, stream>>>(
        qp_f, kp_f, vp_f, rot, trans, qg_bf, kg_bf, vgT_bf, Qsq, Ksq);

    // -------- 5) pair bias GEMM: pair[262144 x 128] @ Wpb[128 x 12] --------
    gemm_Af32Bt(stream, pair, DPAIR, WpbT, DPAIR, pb, NH, bpb, LSEQ * LSEQ, NH, DPAIR);

    // -------- 6) fused logits (QK^T + geometry + bias + mask) --------
    logits_kernel<<<dim3(8, 32, NH), 128, 0, stream>>>(
        q_bf, k_bf, qg_bf, kg_bf, Qsq, Ksq, pb, mask, w_c, w_l, gamma_h, logits);

    // -------- 7) softmax --------
    softmax_kernel<<<NH * LSEQ, 256, 0, stream>>>(logits, attn_bf);

    // -------- 8) attn @ V (seq_out -> combined[:,0:384]) and attn @ Vg (pts) --------
    for (int h = 0; h < NH; ++h) {
        gemm_AbfBt(stream, attn_bf + (long)h * LSEQ * LSEQ, LSEQ,
                   vT_bf + (long)h * DH * LSEQ, LSEQ,
                   nullptr, comb_bf + h * DH, OUTD, nullptr, LSEQ, DH, LSEQ);
        gemm_AbfBt(stream, attn_bf + (long)h * LSEQ * LSEQ, LSEQ,
                   vgT_bf + (long)h * 24 * LSEQ, LSEQ,
                   pts_f + h * 24, nullptr, NH * 24, nullptr, LSEQ, 24, LSEQ);
    }

    // -------- 9) point features (inverse rotation + norm) --------
    pts_feat_kernel<<<(LSEQ * NH * PV + 255) / 256, 256, 0, stream>>>(
        pts_f, rot, trans, comb_bf);

    // -------- 10) attn @ pair (pair_out -> combined[:,768:2304]) --------
    pair_out_kernel<<<LSEQ, 256, 0, stream>>>(attn_bf, pair, comb_bf);

    // -------- 11) final projection: combined[512 x 2304] @ Wo + bo --------
    gemm_AbfBt(stream, comb_bf, OUTD, WoT, OUTD, out, nullptr, DSNG, bo, LSEQ, DSNG, OUTD);
}
